// GraphHead_69784628626298
// MI455X (gfx1250) — matrix operand order repaired and checked by reading, not compiled, not run on hardware
//
#include <hip/hip_runtime.h>
#include <stdint.h>

#define N_NODES 100000
#define N_EDGES 320000
#define HID 256
#define NE 128
#define N_LAYERS 3

typedef __bf16 bf16_t;
typedef bf16_t v16bf __attribute__((ext_vector_type(16)));
typedef float  v8f   __attribute__((ext_vector_type(8)));
typedef unsigned short ushort_t;

static_assert(sizeof(v16bf) == 32, "v16bf must be 32 bytes");

union FragB { uint4 q[2]; v16bf v; };

__device__ __forceinline__ ushort_t f32_to_bf16_rne(float f) {
    unsigned u = __float_as_uint(f);
    u += 0x7FFFu + ((u >> 16) & 1u);
    return (ushort_t)(u >> 16);
}

__device__ __forceinline__ float bf16_bits_lo(unsigned packed) {
    return __uint_as_float(packed << 16);
}
__device__ __forceinline__ float bf16_bits_hi(unsigned packed) {
    return __uint_as_float(packed & 0xFFFF0000u);
}

// gfx1250 async copy: global -> LDS, 16 bytes per lane, tracked by ASYNCcnt.
__device__ __forceinline__ void async_copy_b128(unsigned lds_off, const void* gptr) {
    asm volatile("global_load_async_to_lds_b128 %0, %1, off"
                 :: "v"(lds_off), "v"((unsigned long long)(uintptr_t)gptr)
                 : "memory");
}
__device__ __forceinline__ void wait_async0() {
    asm volatile("s_wait_asynccnt 0" ::: "memory");
}

// ---------------------------------------------------------------------------
// Weight conversion: bf16, K-major (transposed) panels.
//   wsage[l][n][k], k<256 -> sage_Wl[l][k][n], else sage_Wr[l][k-256][n]
//   whead[n][k]         -> head_W0[k][n]
// ---------------------------------------------------------------------------
__global__ void k_convert_w(const float* __restrict__ Wl,
                            const float* __restrict__ Wr,
                            const float* __restrict__ W0,
                            ushort_t* __restrict__ wsage,
                            ushort_t* __restrict__ whead) {
    int idx = blockIdx.x * blockDim.x + threadIdx.x;
    const int SAGE_ELEMS = N_LAYERS * HID * 512;   // 393216
    if (idx < SAGE_ELEMS) {
        int l   = idx / (HID * 512);
        int rem = idx - l * (HID * 512);
        int n   = rem >> 9;
        int k   = rem & 511;
        float v = (k < HID) ? Wl[(size_t)l * HID * HID + (size_t)k * HID + n]
                            : Wr[(size_t)l * HID * HID + (size_t)(k - HID) * HID + n];
        wsage[idx] = f32_to_bf16_rne(v);
    } else {
        int idx2 = idx - SAGE_ELEMS;
        if (idx2 < HID * HID) {
            int n = idx2 >> 8;
            int k = idx2 & 255;
            whead[idx2] = f32_to_bf16_rne(W0[(size_t)k * HID + n]);
        }
    }
}

// ---------------------------------------------------------------------------
// Node encoder -> bf16 features. One 128-thread block per node.
// ---------------------------------------------------------------------------
__global__ __launch_bounds__(128) void k_encode(
        const float* __restrict__ node_attr,
        const float* __restrict__ node_emb,
        const float* __restrict__ net_W, const float* __restrict__ net_b,
        const float* __restrict__ dev_W, const float* __restrict__ dev_b,
        const float* __restrict__ pin_emb,
        const int*   __restrict__ node_type,
        ushort_t* __restrict__ xbf) {
    int n = blockIdx.x;
    int j = threadIdx.x;                 // 0..127
    __shared__ float attr[17];
    if (j < 17) attr[j] = node_attr[(size_t)n * 17 + j];
    __syncthreads();
    int t = node_type[n];
    xbf[(size_t)n * HID + j] = f32_to_bf16_rne(node_emb[t * NE + j]);
    float v = 0.0f;
    if (t == 0) {
        v = net_b[j];
        #pragma unroll
        for (int i = 0; i < 17; ++i) v = fmaf(attr[i], net_W[i * NE + j], v);
    } else if (t == 1) {
        v = dev_b[j];
        #pragma unroll
        for (int i = 0; i < 17; ++i) v = fmaf(attr[i], dev_W[i * NE + j], v);
    } else if (t == 2) {
        int idx = (int)attr[0];
        idx = idx < 0 ? 0 : (idx > 16 ? 16 : idx);
        v = pin_emb[idx * NE + j];
    }
    xbf[(size_t)n * HID + NE + j] = f32_to_bf16_rne(v);
}

// ---------------------------------------------------------------------------
// Helpers
// ---------------------------------------------------------------------------
__global__ void k_zero(float* __restrict__ p, long long n) {
    long long i = (long long)blockIdx.x * blockDim.x + threadIdx.x;
    long long stride = (long long)gridDim.x * blockDim.x;
    for (; i < n; i += stride) p[i] = 0.0f;
}

__global__ void k_degree(const int* __restrict__ dst, float* __restrict__ deg) {
    int e = blockIdx.x * blockDim.x + threadIdx.x;
    if (e < N_EDGES) atomicAdd(&deg[dst[e]], 1.0f);
}

__global__ void k_invdenom(float* __restrict__ deg) {
    int n = blockIdx.x * blockDim.x + threadIdx.x;
    if (n < N_NODES) deg[n] = 1.0f / fmaxf(deg[n], 1.0f);
}

// ---------------------------------------------------------------------------
// Edge scatter: agg[dst] += f32(x_bf16[src]).  4 edges per block,
// 64 lanes/edge, 8-byte bf16 gather (halved random traffic) + 4 f32 atomics.
// ---------------------------------------------------------------------------
__global__ __launch_bounds__(256) void k_scatter(
        const int* __restrict__ src, const int* __restrict__ dst,
        const ushort_t* __restrict__ xbf, float* __restrict__ agg) {
    int le = threadIdx.x >> 6;              // 0..3
    int f  = (threadIdx.x & 63) * 4;        // 0..252
    int e  = blockIdx.x * 4 + le;
    if (e >= N_EDGES) return;
    int s = src[e];
    int d = dst[e];
    uint2 raw = *(const uint2*)&xbf[(size_t)s * HID + f];
    float* ap = &agg[(size_t)d * HID + f];
    atomicAdd(ap + 0, bf16_bits_lo(raw.x));
    atomicAdd(ap + 1, bf16_bits_hi(raw.x));
    atomicAdd(ap + 2, bf16_bits_lo(raw.y));
    atomicAdd(ap + 3, bf16_bits_hi(raw.y));
}

// ---------------------------------------------------------------------------
// WMMA GEMM.  K==512: out = relu([agg*inv_denom | x_bf] @ Wt^T + bias)
//             K==256: out = relu(x_bf @ Wt^T + bias)
// A panel staged into LDS via global_load_async_to_lds_b128:
//   - bf16 x-half copied directly (no conversion needed),
//   - f32 agg-half copied raw, then converted LDS->LDS with 1/deg folded in.
// 8 waves, wave w owns columns [w*32, w*32+32); 16 v_wmma per tile (K=512).
// ---------------------------------------------------------------------------
template<int K>
__global__ __launch_bounds__(256) void k_gemm(
        const float* __restrict__ agg, const float* __restrict__ scale,
        const ushort_t* __restrict__ xbf,
        const ushort_t* __restrict__ Wt,
        const float* __restrict__ bias,
        ushort_t* __restrict__ outbf) {
    constexpr int kshift = (K == 512) ? 9 : 8;
    __shared__ __align__(16) ushort_t sA[16 * K];                 // bf16 A panel
    __shared__ __align__(16) float    sRaw[(K == 512) ? 16 * 256 : 16]; // raw agg
    __shared__ float sScale[16];

    const int rowbase = blockIdx.x * 16;
    const int tid     = threadIdx.x;
    const unsigned sA_off   = (unsigned)(uintptr_t)(void*)sA;
    const unsigned sRaw_off = (unsigned)(uintptr_t)(void*)sRaw;

    if (K == 512) {
        // agg f32 half: 16 rows x 256 cols x 4B = 16KB = 1024 x 16B chunks
        #pragma unroll
        for (int i = 0; i < 4; ++i) {
            int c = tid + i * 256;
            int row = c >> 6, ci = c & 63;
            async_copy_b128(sRaw_off + c * 16,
                            agg + (size_t)(rowbase + row) * HID + ci * 4);
        }
        if (tid < 16) sScale[tid] = scale[rowbase + tid];
    }
    // x bf16 half: 16 rows x 256 cols x 2B = 8KB = 512 x 16B chunks
    #pragma unroll
    for (int i = 0; i < 2; ++i) {
        int c = tid + i * 256;
        int row = c >> 5, ci = c & 31;
        async_copy_b128(sA_off + (unsigned)(row * K + (K - 256) + ci * 8) * 2,
                        xbf + (size_t)(rowbase + row) * HID + ci * 8);
    }
    wait_async0();
    __syncthreads();

    if (K == 512) {
        // convert agg half LDS->LDS with per-row scale
        #pragma unroll
        for (int i = 0; i < 16; ++i) {
            int ei = tid + i * 256;
            int row = ei >> 8, col = ei & 255;
            sA[(row << kshift) + col] = f32_to_bf16_rne(sRaw[ei] * sScale[row]);
        }
        __syncthreads();
    }

    const int lane = tid & 31;
    const int m    = lane & 15;
    const int half = lane >> 4;
    const int wv   = tid >> 5;        // wave 0..7
    const int ctb  = wv * 32;         // column base

    v8f acc0 = {0.f, 0.f, 0.f, 0.f, 0.f, 0.f, 0.f, 0.f};
    v8f acc1 = {0.f, 0.f, 0.f, 0.f, 0.f, 0.f, 0.f, 0.f};

    const uint4* wt0 = (const uint4*)(Wt + (size_t)(ctb + m) * K);
    const uint4* wt1 = (const uint4*)(Wt + (size_t)(ctb + 16 + m) * K);

    const int nkt = K >> 5;
    #pragma unroll 2
    for (int kt = 0; kt < nkt; ++kt) {
        const int k0 = kt << 5;
        // A fragment (ISA 16-bit A layout): two contiguous 16B LDS reads
        FragB a, b0, b1;
        const uint4* pa = (const uint4*)&sA[(m << kshift) + k0 + 8 * half];
        a.q[0] = pa[0];
        a.q[1] = pa[2];
        // B fragment (same K order as A: K=k0+8*half+0..7, then k0+16+8*half+0..7)
        const int bq = (k0 + 8 * half) >> 3;
        b0.q[0] = wt0[bq]; b0.q[1] = wt0[bq + 2];
        b1.q[0] = wt1[bq]; b1.q[1] = wt1[bq + 2];
        if (kt + 1 < nkt) {
            __builtin_prefetch(&wt0[bq + 4], 0, 1);
            __builtin_prefetch(&wt1[bq + 4], 0, 1);
        }
        acc0 = __builtin_amdgcn_wmma_f32_16x16x32_bf16(
                   false, a.v, false, b0.v, (short)0, acc0, false, false);
        acc1 = __builtin_amdgcn_wmma_f32_16x16x32_bf16(
                   false, a.v, false, b1.v, (short)0, acc1, false, false);
    }

    // C/D layout: VGPR r holds M = r + 8*half, N = lane%16 (+tile)
    const int col0 = ctb + m;
    const int col1 = ctb + 16 + m;
    const float bi0 = bias[col0];
    const float bi1 = bias[col1];
    #pragma unroll
    for (int r = 0; r < 8; ++r) {
        int row = rowbase + r + 8 * half;
        outbf[(size_t)row * HID + col0] = f32_to_bf16_rne(fmaxf(acc0[r] + bi0, 0.0f));
        outbf[(size_t)row * HID + col1] = f32_to_bf16_rne(fmaxf(acc1[r] + bi1, 0.0f));
    }
}

// ---------------------------------------------------------------------------
// Head layer 2: pred[n] = dot(h_bf16[n], W1) + b1.  One wave per node.
// ---------------------------------------------------------------------------
__global__ __launch_bounds__(256) void k_head1(
        const ushort_t* __restrict__ hbf, const float* __restrict__ W1,
        const float* __restrict__ b1, float* __restrict__ pred) {
    int wv   = threadIdx.x >> 5;
    int lane = threadIdx.x & 31;
    int n    = blockIdx.x * 8 + wv;
    if (n >= N_NODES) return;
    uint4 hq = *(const uint4*)&hbf[(size_t)n * HID + lane * 8];
    const float4* wp = (const float4*)&W1[lane * 8];
    float4 w0 = wp[0], w1 = wp[1];
    float s = bf16_bits_lo(hq.x) * w0.x + bf16_bits_hi(hq.x) * w0.y
            + bf16_bits_lo(hq.y) * w0.z + bf16_bits_hi(hq.y) * w0.w
            + bf16_bits_lo(hq.z) * w1.x + bf16_bits_hi(hq.z) * w1.y
            + bf16_bits_lo(hq.w) * w1.z + bf16_bits_hi(hq.w) * w1.w;
    #pragma unroll
    for (int off = 16; off > 0; off >>= 1) s += __shfl_down(s, off, 32);
    if (lane == 0) pred[n] = s + b1[0];
}

// ---------------------------------------------------------------------------
// Tail: true_class (int32 bits) and true_label passthrough.
// d_out layout: [pred N | true_class N (int32) | true_label N*2]
// ---------------------------------------------------------------------------
__global__ void k_tail(const float* __restrict__ y, float* __restrict__ out) {
    int n = blockIdx.x * blockDim.x + threadIdx.x;
    if (n >= N_NODES) return;
    float y0 = y[(size_t)n * 2 + 0];
    float y1 = y[(size_t)n * 2 + 1];
    ((int*)out)[N_NODES + n] = (int)y1;
    out[(size_t)2 * N_NODES + 2 * n + 0] = y0;
    out[(size_t)2 * N_NODES + 2 * n + 1] = y1;
}

// ---------------------------------------------------------------------------
extern "C" void kernel_launch(void* const* d_in, const int* in_sizes, int n_in,
                              void* d_out, int out_size, void* d_ws, size_t ws_size,
                              hipStream_t stream) {
    const float* node_attr = (const float*)d_in[0];
    const float* y         = (const float*)d_in[1];
    const float* node_emb  = (const float*)d_in[2];
    // d_in[3] edge_emb: unused by SAGE (reference discards xe)
    const float* net_W     = (const float*)d_in[4];
    const float* net_b     = (const float*)d_in[5];
    const float* dev_W     = (const float*)d_in[6];
    const float* dev_b     = (const float*)d_in[7];
    const float* pin_emb   = (const float*)d_in[8];
    const float* sage_Wl   = (const float*)d_in[9];
    const float* sage_bl   = (const float*)d_in[10];
    const float* sage_Wr   = (const float*)d_in[11];
    const float* head_W0   = (const float*)d_in[12];
    const float* head_b0   = (const float*)d_in[13];
    const float* head_W1   = (const float*)d_in[14];
    const float* head_b1   = (const float*)d_in[15];
    const int*   node_type = (const int*)d_in[16];
    // d_in[17] edge_type: unused
    const int*   edge_index = (const int*)d_in[18];
    const int*   esrc = edge_index;
    const int*   edst = edge_index + N_EDGES;
    float* out = (float*)d_out;

    // Workspace carve-up (~156 MB).
    char* ws = (char*)d_ws;
    size_t off = 0;
    ushort_t* xbf = (ushort_t*)(ws + off); off += (size_t)N_NODES * HID * sizeof(ushort_t);
    float*    agg = (float*)(ws + off);    off += (size_t)N_NODES * HID * sizeof(float);
    float*    deg = (float*)(ws + off);    off += (size_t)N_NODES * sizeof(float);
    off = (off + 255) & ~(size_t)255;
    ushort_t* wsage = (ushort_t*)(ws + off);
    off += (size_t)N_LAYERS * HID * 512 * sizeof(ushort_t);
    ushort_t* whead = (ushort_t*)(ws + off);
    ushort_t* hbf   = (ushort_t*)agg;      // head activations reuse agg buffer

    // 1) weights -> bf16 (transposed, K-major)
    k_convert_w<<<(N_LAYERS * HID * 512 + HID * HID + 255) / 256, 256, 0, stream>>>(
        sage_Wl, sage_Wr, head_W0, wsage, whead);

    // 2) node encoder -> bf16 features
    k_encode<<<N_NODES, 128, 0, stream>>>(node_attr, node_emb, net_W, net_b,
                                          dev_W, dev_b, pin_emb, node_type, xbf);

    // 3) degree -> 1/max(deg,1)
    k_zero<<<512, 256, 0, stream>>>(deg, (long long)N_NODES);
    k_degree<<<(N_EDGES + 255) / 256, 256, 0, stream>>>(edst, deg);
    k_invdenom<<<(N_NODES + 255) / 256, 256, 0, stream>>>(deg);

    // 4) three SAGE layers: zero agg -> scatter -> fused WMMA GEMM (in-place x)
    for (int l = 0; l < N_LAYERS; ++l) {
        k_zero<<<2048, 256, 0, stream>>>(agg, (long long)N_NODES * HID);
        k_scatter<<<(N_EDGES + 3) / 4, 256, 0, stream>>>(esrc, edst, xbf, agg);
        k_gemm<512><<<N_NODES / 16, 256, 0, stream>>>(
            agg, deg, xbf, wsage + (size_t)l * HID * 512,
            sage_bl + (size_t)l * HID, xbf);
    }

    // 5) head: h = relu(x @ W0 + b0) via WMMA, then per-node dot with W1
    k_gemm<256><<<N_NODES / 16, 256, 0, stream>>>(
        nullptr, nullptr, xbf, whead, head_b0, hbf);
    k_head1<<<N_NODES / 8, 256, 0, stream>>>(hbf, head_W1, head_b1, out);

    // 6) passthrough outputs
    k_tail<<<(N_NODES + 255) / 256, 256, 0, stream>>>(y, out);
}
